// peepLSTM_87857851007790
// MI455X (gfx1250) — matrix operand order repaired and checked by reading, compiled, not run on hardware
//
#include <hip/hip_runtime.h>
#include <hip/hip_bf16.h>

// Peephole LSTM for MI455X (gfx1250, wave32, WMMA).
//
// x in {0,1,2} => all four x-projections collapse to 3-row lookup tables
// G[g][v][h] = emb[v] @ w_gx + b_g (pre-sigmoided for the c-gate).  The heavy
// compute is the sequential recurrence (i,f gates; o only on the last step):
// ~550 GFLOP of bf16 WMMA over 511 strictly-dependent steps.  Weights (6MB
// bf16, pre-swizzled to B-fragment order) and c (1MB) are L2-resident; one
// persistent kernel with a software grid barrier performs all steps.
// c is staged per-block into LDS in per-lane A-fragment order so the K-loop
// is 2x ds_load_b128 + 4x global_load_b128 + 2x v_wmma per K-tile.
// An opaque (asm) zero byte-offset added to the weight base each step stops
// the compiler hoisting the step-invariant B-loads out of the s-loop (which
// would spill to scratch) while keeping pointer provenance rooted at the
// global kernarg => global_load with saddr, not flat.

#define B_ 256
#define S_ 512
#define H_ 1024
#define E_ 512
#define C_ 10
#define NBLOCKS 128   // 8 M-tiles (32 rows) x 16 N-tiles (64 cols)

typedef __attribute__((ext_vector_type(16))) __bf16 v16bf;
typedef __attribute__((ext_vector_type(8)))  float  v8f;

union FragU {
  v16bf v;
  unsigned int u32[8];
  uint4 u4[2];
};

__device__ __forceinline__ unsigned short f2bf(float f) {
  unsigned int u = __float_as_uint(f);
  unsigned int r = u + 0x7FFFu + ((u >> 16) & 1u);   // round-to-nearest-even
  return (unsigned short)(r >> 16);
}

// fast sigmoid: v_exp_f32 + v_rcp_f32 (avoid the v_div_scale IEEE-div macro)
__device__ __forceinline__ float sigm_fast(float z) {
  return __builtin_amdgcn_rcpf(1.0f + __expf(-z));
}

__device__ __forceinline__ float sel3(float a0, float a1, float a2, int v) {
  float r = (v == 1) ? a1 : a0;
  return (v == 2) ? a2 : r;
}

// ---------------------------------------------------------------- init
__global__ void k_init(float* c0, unsigned int* bar) {
  int i = blockIdx.x * 256 + threadIdx.x;
  if (i < B_ * H_) c0[i] = 0.0f;
  if (i < 2) bar[i] = 0u;   // arrival counter + generation, every call (determinism)
}

// ------------------------------------------------- gate lookup tables
// Gt[(g*3+v)*H + h] = b_g[h] + sum_e emb[v][e] * w_gx[e][h];  g: 0=i,1=f,2=o,3=c
// The c-gate table is stored pre-sigmoided (only ever used through sigmoid).
__global__ void k_gates(const float* __restrict__ emb,
                        const float* __restrict__ wix, const float* __restrict__ wfx,
                        const float* __restrict__ wox, const float* __restrict__ wcx,
                        const float* __restrict__ bi,  const float* __restrict__ bf,
                        const float* __restrict__ bo,  const float* __restrict__ bc,
                        float* __restrict__ Gt) {
  int h = blockIdx.x * 256 + threadIdx.x;   // gridDim.x = H/256
  int v = blockIdx.y;                       // 0..2
  int g = blockIdx.z;                       // 0..3
  const float* W = (g == 0) ? wix : (g == 1) ? wfx : (g == 2) ? wox : wcx;
  const float* bb = (g == 0) ? bi : (g == 1) ? bf : (g == 2) ? bo : bc;
  float acc = bb[h];
  for (int e = 0; e < E_; ++e) acc += emb[v * E_ + e] * W[e * H_ + h];
  if (g == 3) acc = 1.0f / (1.0f + expf(-acc));   // pre-sigmoided c-gate
  Gt[(g * 3 + v) * H_ + h] = acc;
}

// ------------------------------- pre-swizzle recurrent weights to bf16
// Packed per matrix as [nt(64)][kt(32)][lane(32)][slot(16)] so each lane in the
// hot loop loads its 16 B-fragment bf16 values with two coalesced b128 loads.
// B-fragment k order used here (contiguous, same as the A staging; 32x16 KxN):
//   lane<16 : column N=lane,    K = slot       (0..15)
//   lane>=16: column N=lane-16, K = 16 + slot  (16..31)
__global__ void k_swizzle(const float* __restrict__ wih,
                          const float* __restrict__ wfh,
                          const float* __restrict__ woh,
                          unsigned short* __restrict__ Wp) {
  int tid = blockIdx.x * 256 + threadIdx.x;
  const int per = H_ * H_;
  int mat = tid / per;
  int r = tid - mat * per;
  int slot = r & 15;
  int lane = (r >> 4) & 31;
  int kt   = (r >> 9) & 31;
  int nt   = r >> 14;
  int kk = (lane < 16) ? slot : (16 + slot);
  int n = nt * 16 + (lane & 15);
  int k = kt * 32 + kk;
  const float* W = (mat == 0) ? wih : (mat == 1) ? wfh : woh;
  Wp[tid] = f2bf(W[k * H_ + n]);
}

// ---------------------------------------------------------- grid barrier
__device__ __forceinline__ void grid_sync(unsigned int* bar) {
  __syncthreads();
  __threadfence();
  if (threadIdx.x == 0) {
    unsigned int* cnt = bar;
    unsigned int* gen = bar + 1;
    unsigned int g = __hip_atomic_load(gen, __ATOMIC_RELAXED, __HIP_MEMORY_SCOPE_AGENT);
    unsigned int a = __hip_atomic_fetch_add(cnt, 1u, __ATOMIC_ACQ_REL, __HIP_MEMORY_SCOPE_AGENT);
    if (a == NBLOCKS - 1u) {
      __hip_atomic_store(cnt, 0u, __ATOMIC_RELAXED, __HIP_MEMORY_SCOPE_AGENT);
      __hip_atomic_store(gen, g + 1u, __ATOMIC_RELEASE, __HIP_MEMORY_SCOPE_AGENT);
    } else {
      while (__hip_atomic_load(gen, __ATOMIC_ACQUIRE, __HIP_MEMORY_SCOPE_AGENT) == g) {
        __builtin_amdgcn_s_sleep(2);
      }
    }
  }
  __syncthreads();
}

// ---------------------------------------------------------- persistent LSTM
__global__ __launch_bounds__(256) void k_lstm(
    const long long* __restrict__ x,
    const float* __restrict__ Gt,
    const unsigned short* __restrict__ Wp,
    float* __restrict__ cb0, float* __restrict__ cb1,
    float* __restrict__ hbuf,
    const float* __restrict__ w_ph, const float* __restrict__ b_p,
    float* __restrict__ out,
    unsigned int* __restrict__ bar) {

  // LDS: c tile staged directly in A-fragment order:
  //   dword index = ((waveM*32 + kt)*32 + lane)*8 + r   (pair r of the frag)
  __shared__ union {
    unsigned int a[2 * 32 * 32 * 8];   // 16384 dwords = 64 KB
    float proj[2570];                  // block 0: projection + logsumexp
  } sh;

  const int tid   = threadIdx.x;
  const int lane  = tid & 31;
  const int wv    = tid >> 5;       // 0..7 waves
  const int waveM = wv >> 2;        // 0..1  (16 rows each)
  const int waveN = wv & 3;         // 0..3  (16 cols each)
  const int mb = blockIdx.x >> 4;   // 0..7  -> 32 batch rows
  const int nb = blockIdx.x & 15;   // 0..15 -> 64 h cols
  const int rowBlockBase = mb * 32;
  const int nt_g = nb * 4 + waveN;              // global 16-col tile (0..63)
  const int hcol = nt_g * 16 + (lane & 15);     // D-frag column
  const int hi8  = (lane >= 16) ? 8 : 0;        // D-frag row offset
  const int mloc_base = waveM * 16 + hi8;       // + v (0..7) = local row for D

  float* bufs[2] = {cb0, cb1};

  // Loop-invariant gate-table values: 4 gates x 3 input symbols per lane.
  const float gI0 = Gt[0 * H_ + hcol], gI1 = Gt[1 * H_ + hcol], gI2 = Gt[2 * H_ + hcol];
  const float gF0 = Gt[3 * H_ + hcol], gF1 = Gt[4 * H_ + hcol], gF2 = Gt[5 * H_ + hcol];
  const float gO0 = Gt[6 * H_ + hcol], gO1 = Gt[7 * H_ + hcol], gO2 = Gt[8 * H_ + hcol];
  const float gC0 = Gt[9 * H_ + hcol], gC1 = Gt[10 * H_ + hcol], gC2 = Gt[11 * H_ + hcol];

  // per-lane element offset of this wave's B-fragment strip within a matrix
  const int bOffElems = (nt_g * 32 * 32 + lane) * 16;   // divergent (32-bit voffset)

  // stage 32 rows x 1024 cols of c (fp32 global -> bf16 LDS, A-fragment order)
  // group = 4 consecutive bf16 pairs (always contiguous slots r0..r0+3 of one
  // lane's fragment) -> one 16B LDS store per group.
  auto stage = [&](const float* cc) {
    for (int j = 0; j < 16; ++j) {
      int g4 = tid + 256 * j;          // group index 0..4095
      int m  = g4 >> 7;                // row 0..31
      int cg = g4 & 127;               // group within row (4 pairs = 8 floats)
      const float4* src = (const float4*)(cc + (size_t)(rowBlockBase + m) * H_ + cg * 8);
      float4 f0 = src[0];
      float4 f1 = src[1];
      uint4 pk;
      pk.x = (unsigned int)f2bf(f0.x) | ((unsigned int)f2bf(f0.y) << 16);
      pk.y = (unsigned int)f2bf(f0.z) | ((unsigned int)f2bf(f0.w) << 16);
      pk.z = (unsigned int)f2bf(f1.x) | ((unsigned int)f2bf(f1.y) << 16);
      pk.w = (unsigned int)f2bf(f1.z) | ((unsigned int)f2bf(f1.w) << 16);
      int w4 = (cg & 3) * 4;                         // 0,4,8,12 within 16-pair tile
      int kt = cg >> 2;                              // K-tile 0..31
      int laneDst = (m & 15) + ((w4 & 8) ? 16 : 0);  // frag lane
      int r0 = w4 & 4;                               // frag pair slot 0 or 4
      int wM = m >> 4;
      *(uint4*)&sh.a[(((wM * 32 + kt) * 32 + laneDst) * 8 + r0)] = pk;
    }
  };

  // A fragment: 2 contiguous b128 LDS loads, address linear in kt
  auto loadA = [&](int kt) {
    FragU a;
    const uint4* pA = (const uint4*)&sh.a[((waveM * 32 + kt) * 32 + lane) * 8];
    a.u4[0] = pA[0];
    a.u4[1] = pA[1];
    return a;
  };

  // B fragment from a packed-weight strip base; address linear in kt
  auto loadB = [&](const unsigned short* strip, int kt) {
    FragU b;
    const uint4* p4 = (const uint4*)(strip + (size_t)kt * (32 * 16));
    b.u4[0] = p4[0];
    b.u4[1] = p4[1];
    return b;
  };

  // ----------------- main recurrence: steps 0 .. S-3 (i,f gates only) ------
  for (int s = 0; s < S_ - 2; ++s) {
    const float* cc = bufs[s & 1];
    float* cnx = bufs[(s & 1) ^ 1];

    stage(cc);
    __syncthreads();

    // Opaque zero offset (SGPR) regenerated each step: the B-load addresses
    // are not loop-invariant (no hoist/spill across 510 steps), but pointer
    // provenance stays rooted at the global kernarg => global_load + saddr.
    size_t wSkew = 0;
    asm volatile("" : "+s"(wSkew));
    const unsigned short* wI = Wp + wSkew + bOffElems;
    const unsigned short* wF = wI + H_ * H_;

    v8f accI = {0, 0, 0, 0, 0, 0, 0, 0};
    v8f accF = {0, 0, 0, 0, 0, 0, 0, 0};
#pragma unroll 8
    for (int kt = 0; kt < 32; ++kt) {
      FragU a  = loadA(kt);
      FragU bi = loadB(wI, kt);
      FragU bf = loadB(wF, kt);
      accI = __builtin_amdgcn_wmma_f32_16x16x32_bf16(false, a.v, false, bi.v,
                                                     (short)0, accI, false, false);
      accF = __builtin_amdgcn_wmma_f32_16x16x32_bf16(false, a.v, false, bf.v,
                                                     (short)0, accF, false, false);
    }

#pragma unroll
    for (int v = 0; v < 8; ++v) {
      int b = rowBlockBase + mloc_base + v;
      int xv = (int)x[(size_t)b * S_ + s];
      float gi  = sel3(gI0, gI1, gI2, xv);
      float gf  = sel3(gF0, gF1, gF2, xv);
      float gcs = sel3(gC0, gC1, gC2, xv);           // already sigmoided
      float cold = cc[(size_t)b * H_ + hcol];
      float cnew = gcs * sigm_fast(accI[v] + gi) + cold * sigm_fast(accF[v] + gf);
      cnx[(size_t)b * H_ + hcol] = cnew;
    }
    grid_sync(bar);
  }

  // ----------------- final step s = S-2: i,f,o gates + h --------------------
  {
    const int s = S_ - 2;
    const float* cc = bufs[s & 1];

    stage(cc);
    __syncthreads();

    size_t wSkew = 0;
    asm volatile("" : "+s"(wSkew));
    const unsigned short* wI = Wp + wSkew + bOffElems;
    const unsigned short* wF = wI + H_ * H_;
    const unsigned short* wO = wF + H_ * H_;

    v8f accI = {0, 0, 0, 0, 0, 0, 0, 0};
    v8f accF = {0, 0, 0, 0, 0, 0, 0, 0};
    v8f accO = {0, 0, 0, 0, 0, 0, 0, 0};
#pragma unroll 4
    for (int kt = 0; kt < 32; ++kt) {
      FragU a  = loadA(kt);
      FragU bi = loadB(wI, kt);
      FragU bf = loadB(wF, kt);
      FragU bo = loadB(wO, kt);
      accI = __builtin_amdgcn_wmma_f32_16x16x32_bf16(false, a.v, false, bi.v,
                                                     (short)0, accI, false, false);
      accF = __builtin_amdgcn_wmma_f32_16x16x32_bf16(false, a.v, false, bf.v,
                                                     (short)0, accF, false, false);
      accO = __builtin_amdgcn_wmma_f32_16x16x32_bf16(false, a.v, false, bo.v,
                                                     (short)0, accO, false, false);
    }

#pragma unroll
    for (int v = 0; v < 8; ++v) {
      int b = rowBlockBase + mloc_base + v;
      int xv = (int)x[(size_t)b * S_ + s];
      float gi  = sel3(gI0, gI1, gI2, xv);
      float gf  = sel3(gF0, gF1, gF2, xv);
      float go  = sel3(gO0, gO1, gO2, xv);
      float gcs = sel3(gC0, gC1, gC2, xv);
      float cold = cc[(size_t)b * H_ + hcol];
      float iv = sigm_fast(accI[v] + gi);
      float fv = sigm_fast(accF[v] + gf);
      float ov = sigm_fast(accO[v] + go);
      float cnew = gcs * iv + cold * fv;
      float pv = (float)((xv + 1) >> 1);     // pad = (x+1)//2
      hbuf[(size_t)b * H_ + hcol] = tanhf(cnew) * ov * pv;
    }
    grid_sync(bar);
  }

  // ----------------- projection + log_softmax over batch (axis 0) ----------
  if (blockIdx.x == 0) {
    const int b = tid;   // 0..255
    float acc[C_];
#pragma unroll
    for (int c = 0; c < C_; ++c) acc[c] = b_p[c];
    for (int k = 0; k < H_; ++k) {
      float hv = hbuf[(size_t)b * H_ + k];
#pragma unroll
      for (int c = 0; c < C_; ++c) acc[c] += hv * w_ph[k * C_ + c];
    }
#pragma unroll
    for (int c = 0; c < C_; ++c) sh.proj[b * C_ + c] = acc[c];
    __syncthreads();
    if (tid < C_) {
      float m = -1e30f;
      for (int bb = 0; bb < B_; ++bb) m = fmaxf(m, sh.proj[bb * C_ + tid]);
      float ssum = 0.0f;
      for (int bb = 0; bb < B_; ++bb) ssum += __expf(sh.proj[bb * C_ + tid] - m);
      sh.proj[2560 + tid] = m + __logf(ssum);
    }
    __syncthreads();
#pragma unroll
    for (int c = 0; c < C_; ++c)
      out[b * C_ + c] = sh.proj[b * C_ + c] - sh.proj[2560 + c];
  }
}

// ---------------------------------------------------------------- launcher
extern "C" void kernel_launch(void* const* d_in, const int* in_sizes, int n_in,
                              void* d_out, int out_size, void* d_ws, size_t ws_size,
                              hipStream_t stream) {
  (void)in_sizes; (void)n_in; (void)out_size; (void)ws_size;

  const long long* x = (const long long*)d_in[0];
  const float* emb  = (const float*)d_in[1];
  const float* w_cx = (const float*)d_in[2];
  const float* b_c  = (const float*)d_in[3];
  const float* w_ix = (const float*)d_in[4];
  const float* w_ih = (const float*)d_in[5];
  const float* b_i  = (const float*)d_in[6];
  const float* w_fx = (const float*)d_in[7];
  const float* w_fh = (const float*)d_in[8];
  const float* b_f  = (const float*)d_in[9];
  const float* w_ox = (const float*)d_in[10];
  const float* w_oh = (const float*)d_in[11];
  const float* b_o  = (const float*)d_in[12];
  const float* w_ph = (const float*)d_in[13];
  const float* b_p  = (const float*)d_in[14];

  // workspace layout (~9.5 MB)
  char* ws = (char*)d_ws;
  float* Gt   = (float*)ws;                                       // 48 KB
  float* cb0  = (float*)(ws + (64 << 10));                        // 1 MB
  float* cb1  = (float*)(ws + (64 << 10) + (1 << 20));            // 1 MB
  float* hbuf = (float*)(ws + (64 << 10) + (2 << 20));            // 1 MB
  unsigned short* Wp = (unsigned short*)(ws + (64 << 10) + (3 << 20));  // 6 MB
  unsigned int* bar  = (unsigned int*)(ws + (64 << 10) + (9 << 20));    // 8 B

  k_init<<<(B_ * H_ + 255) / 256, 256, 0, stream>>>(cb0, bar);

  dim3 gg(H_ / 256, 3, 4);
  k_gates<<<gg, 256, 0, stream>>>(emb, w_ix, w_fx, w_ox, w_cx,
                                  b_i, b_f, b_o, b_c, Gt);

  k_swizzle<<<(3 * H_ * H_) / 256, 256, 0, stream>>>(w_ih, w_fh, w_oh, Wp);

  k_lstm<<<NBLOCKS, 256, 0, stream>>>(x, Gt, Wp, cb0, cb1, hbuf,
                                      w_ph, b_p, (float*)d_out, bar);
}
